// RaftModule_44976897524425
// MI455X (gfx1250) — matrix operand order repaired and checked by reading, not compiled, not run on hardware
//
#include <hip/hip_runtime.h>
#include <hip/hip_bf16.h>
#include <math.h>

// =====================================================================
// RAFT forward for MI455X (gfx1250, wave32).
// Conv / correlation GEMMs on v_wmma_f32_16x16x32_f16 with pre-packed
// f16 operands: weights [Cout16][KHKW*Cinp] (K-order = rs-major, ci-minor,
// zero padded), activations channels-last f16 with 3-pixel zero halo.
// Inner loop = 6x global_load_b128 + 2x WMMA per 32-deep K chunk,
// no conditionals, no converts.
// =====================================================================

typedef __attribute__((ext_vector_type(16))) _Float16 v16h;
typedef __attribute__((ext_vector_type(8)))  _Float16 v8h;
typedef __attribute__((ext_vector_type(8)))  float    v8f;

#define EW_BLOCK 256
#define HALO 3

__device__ __forceinline__ int iclamp(int v, int lo, int hi)
{
    return v < lo ? lo : (v > hi ? hi : v);
}

// ---------------------------------------------------------------------
// Weight pack: OIHW f32 -> [Cout16][khkw*Cinp] f16, element (m, rs, ci),
// zero padded in m and ci.
// ---------------------------------------------------------------------
__global__ void pack_weight_kernel(const float* __restrict__ src, _Float16* __restrict__ dst,
                                   int Cout, int Cin, int khkw, int Cinp, int Cout16)
{
    long i = (long)blockIdx.x * blockDim.x + threadIdx.x;
    long total = (long)Cout16 * khkw * Cinp;
    if (i >= total) return;
    int ci = (int)(i % Cinp);
    long t = i / Cinp;
    int rs = (int)(t % khkw);
    int m  = (int)(t / khkw);
    float v = 0.f;
    if (m < Cout && ci < Cin)
        v = src[((size_t)m * Cin + ci) * khkw + rs];
    dst[i] = (_Float16)v;
}

// ---------------------------------------------------------------------
// Activation shadow: NCHW f32 -> [B][(H+6)*(W+6)][Cinp] f16, zero halo,
// zero channel padding.
// ---------------------------------------------------------------------
__global__ void shadow_kernel(const float* __restrict__ src, _Float16* __restrict__ dst,
                              int B, int Cin, int Cinp, int Hin, int Win)
{
    const int Hs = Hin + 2 * HALO, Ws = Win + 2 * HALO;
    long i = (long)blockIdx.x * blockDim.x + threadIdx.x;
    long total = (long)B * Hs * Ws * Cinp;
    if (i >= total) return;
    int c = (int)(i % Cinp);
    long t = i / Cinp;
    int xs = (int)(t % Ws);
    t /= Ws;
    int ys = (int)(t % Hs);
    int b  = (int)(t / Hs);
    int iy = ys - HALO, ix = xs - HALO;
    float v = 0.f;
    if (c < Cin && iy >= 0 && iy < Hin && ix >= 0 && ix < Win)
        v = src[((size_t)(b * Cin + c) * Hin + iy) * Win + ix];
    dst[i] = (_Float16)v;
}

// NCHW f32 -> [B][HW][C] f16 (for correlation features; C multiple of 32)
__global__ void chw_to_hwc_f16_kernel(const float* __restrict__ src, _Float16* __restrict__ dst,
                                      int B, int C, int HW)
{
    long i = (long)blockIdx.x * blockDim.x + threadIdx.x;
    long total = (long)B * HW * C;
    if (i >= total) return;
    int c = (int)(i % C);
    long t = i / C;
    int hw = (int)(t % HW);
    int b  = (int)(t / HW);
    dst[i] = (_Float16)src[((size_t)b * C + c) * HW + hw];
}

// ---------------------------------------------------------------------
// Implicit-GEMM convolution, one wave -> 16(Cout) x 32(pixel) tile.
// xs: f16 shadow [B][Hs*Ws][Cinp]; wp: f16 packed [Cout16][KH*KW*Cinp].
// act: 0 none, 1 relu, 2 sigmoid, 3 tanh
// ---------------------------------------------------------------------
template<int KH, int KW>
__global__ void conv2d_wmma_kernel(const _Float16* __restrict__ xs, const _Float16* __restrict__ wp,
                                   const float* __restrict__ bias, float* __restrict__ y,
                                   int B, int Cinp, int Hin, int Win,
                                   int Cout, int Hout, int Wout,
                                   int stride, int act)
{
    const int lane  = threadIdx.x & 31;
    const int wid   = threadIdx.x >> 5;
    const int wpb   = blockDim.x >> 5;
    const int Npix  = Hout * Wout;
    const int mTiles = (Cout + 15) >> 4;
    const int nTiles = (Npix + 31) >> 5;
    int tile = blockIdx.x * wpb + wid;
    if (tile >= B * mTiles * nTiles) return;
    int b  = tile / (mTiles * nTiles);
    int r  = tile - b * (mTiles * nTiles);
    int mT = r / nTiles;
    int nT = r - mT * nTiles;

    const int Hs = Hin + 2 * HALO, Ws = Win + 2 * HALO;
    const int Kp = KH * KW * Cinp;
    const int baseA = (lane < 16) ? 0 : 8;    // A fragment K group (ISA 7.12.2)
    const int baseB = (lane < 16) ? 0 : 8;    // B fragment K group
    const int mRow  = mT * 16 + (lane & 15);  // always < Cout16 (padded rows)
    const int n0 = nT * 32 + (lane & 15);
    const int n1 = n0 + 16;
    const int n0c = (n0 < Npix) ? n0 : Npix - 1;
    const int n1c = (n1 < Npix) ? n1 : Npix - 1;
    const int oy0 = n0c / Wout, ox0 = n0c - oy0 * Wout;
    const int oy1 = n1c / Wout, ox1 = n1c - oy1 * Wout;
    // shadow coords: +HALO shift makes every tap in-bounds
    const int py0 = oy0 * stride - (KH >> 1) + HALO, px0 = ox0 * stride - (KW >> 1) + HALO;
    const int py1 = oy1 * stride - (KH >> 1) + HALO, px1 = ox1 * stride - (KW >> 1) + HALO;

    const _Float16* wrow = wp + (size_t)mRow * Kp + baseA;
    const _Float16* xb   = xs + (size_t)b * Hs * Ws * Cinp + baseB;

    v8f acc0 = {}, acc1 = {};
#pragma unroll
    for (int rs = 0; rs < KH * KW; ++rs) {
        const int rr = rs / KW, ss = rs - (rs / KW) * KW;
        const _Float16* wk = wrow + (size_t)rs * Cinp;
        const _Float16* x0 = xb + ((size_t)(py0 + rr) * Ws + (px0 + ss)) * Cinp;
        const _Float16* x1 = xb + ((size_t)(py1 + rr) * Ws + (px1 + ss)) * Cinp;
        for (int c0 = 0; c0 < Cinp; c0 += 32) {
            v8h alo = *(const v8h*)(wk + c0);
            v8h ahi = *(const v8h*)(wk + c0 + 16);
            v8h bl0 = *(const v8h*)(x0 + c0);
            v8h bh0 = *(const v8h*)(x0 + c0 + 16);
            v8h bl1 = *(const v8h*)(x1 + c0);
            v8h bh1 = *(const v8h*)(x1 + c0 + 16);
            v16h a, b0, b1;
#pragma unroll
            for (int e = 0; e < 8; ++e) {
                a[e] = alo[e];  a[8 + e] = ahi[e];
                b0[e] = bl0[e]; b0[8 + e] = bh0[e];
                b1[e] = bl1[e]; b1[8 + e] = bh1[e];
            }
            acc0 = __builtin_amdgcn_wmma_f32_16x16x32_f16(false, a, false, b0,
                                                          (short)0, acc0, false, false);
            acc1 = __builtin_amdgcn_wmma_f32_16x16x32_f16(false, a, false, b1,
                                                          (short)0, acc1, false, false);
        }
    }

    const bool mfull = ((Cout & 15) == 0);
#pragma unroll
    for (int t = 0; t < 2; ++t) {
        int n = (t == 0) ? n0 : n1;
        if (n >= Npix) continue;
        const v8f& acc = (t == 0) ? acc0 : acc1;
        int yy = n / Wout, xx = n - (n / Wout) * Wout;
        int mBase = mT * 16 + ((lane < 16) ? 0 : 8);
        float* yp = y + ((size_t)b * Cout * Hout + yy) * Wout + xx;
        if (mfull) {
#pragma unroll
            for (int v = 0; v < 8; ++v) {
                int m = mBase + v;
                float o = acc[v] + bias[m];
                if      (act == 1) o = fmaxf(o, 0.f);
                else if (act == 2) o = 1.f / (1.f + __expf(-o));
                else if (act == 3) o = tanhf(o);
                yp[(size_t)m * Hout * Wout] = o;
            }
        } else {
#pragma unroll
            for (int v = 0; v < 8; ++v) {
                int m = mBase + v;
                if (m < Cout) {
                    float o = acc[v] + bias[m];
                    if      (act == 1) o = fmaxf(o, 0.f);
                    else if (act == 2) o = 1.f / (1.f + __expf(-o));
                    else if (act == 3) o = tanhf(o);
                    yp[(size_t)m * Hout * Wout] = o;
                }
            }
        }
    }
}

// ---------------------------------------------------------------------
// All-pairs correlation on transposed f16 features [B][Np][C].
// corr[b,p,q] = scale * sum_c f1[b,p,c] f2[b,q,c]. Np%32==0, C%32==0.
// ---------------------------------------------------------------------
__global__ void corr_wmma_kernel(const _Float16* __restrict__ f1t, const _Float16* __restrict__ f2t,
                                 float* __restrict__ corr, int B, int C, int Np, float scale)
{
    const int lane = threadIdx.x & 31;
    const int wid  = threadIdx.x >> 5;
    const int wpb  = blockDim.x >> 5;
    const int tM   = Np >> 4;
    const int tN   = Np >> 5;
    int tile = blockIdx.x * wpb + wid;
    if (tile >= B * tM * tN) return;
    int b  = tile / (tM * tN);
    int r  = tile - b * (tM * tN);
    int mT = r / tN, nT = r - (r / tN) * tN;

    const int baseA = (lane < 16) ? 0 : 8;
    const int baseB = (lane < 16) ? 0 : 8;
    const int p  = mT * 16 + (lane & 15);
    const int q0 = nT * 32 + (lane & 15);
    const int q1 = q0 + 16;
    const _Float16* arow = f1t + ((size_t)b * Np + p)  * C + baseA;
    const _Float16* br0  = f2t + ((size_t)b * Np + q0) * C + baseB;
    const _Float16* br1  = f2t + ((size_t)b * Np + q1) * C + baseB;

    v8f acc0 = {}, acc1 = {};
    for (int k0 = 0; k0 < C; k0 += 32) {
        v8h alo = *(const v8h*)(arow + k0);
        v8h ahi = *(const v8h*)(arow + k0 + 16);
        v8h bl0 = *(const v8h*)(br0 + k0);
        v8h bh0 = *(const v8h*)(br0 + k0 + 16);
        v8h bl1 = *(const v8h*)(br1 + k0);
        v8h bh1 = *(const v8h*)(br1 + k0 + 16);
        v16h a, b0, b1;
#pragma unroll
        for (int e = 0; e < 8; ++e) {
            a[e] = alo[e];  a[8 + e] = ahi[e];
            b0[e] = bl0[e]; b0[8 + e] = bh0[e];
            b1[e] = bl1[e]; b1[8 + e] = bh1[e];
        }
        acc0 = __builtin_amdgcn_wmma_f32_16x16x32_f16(false, a, false, b0,
                                                      (short)0, acc0, false, false);
        acc1 = __builtin_amdgcn_wmma_f32_16x16x32_f16(false, a, false, b1,
                                                      (short)0, acc1, false, false);
    }
#pragma unroll
    for (int t = 0; t < 2; ++t) {
        int q = (t == 0) ? q0 : q1;
        const v8f& acc = (t == 0) ? acc0 : acc1;
#pragma unroll
        for (int v = 0; v < 8; ++v) {
            int m = mT * 16 + v + ((lane < 16) ? 0 : 8);
            corr[((size_t)b * Np + m) * Np + q] = acc[v] * scale;
        }
    }
}

// ---------------------------------------------------------------------
// InstanceNorm (affine=False), in-place, optional fused relu.
// ---------------------------------------------------------------------
__global__ void instnorm_kernel(float* __restrict__ data, int HW, int relu)
{
    float* p = data + (size_t)blockIdx.x * HW;
    __shared__ float s1[EW_BLOCK], s2[EW_BLOCK];
    float a = 0.f, b = 0.f;
    for (int i = threadIdx.x; i < HW; i += blockDim.x) { float v = p[i]; a += v; b += v * v; }
    s1[threadIdx.x] = a; s2[threadIdx.x] = b;
    __syncthreads();
    for (int s = blockDim.x >> 1; s > 0; s >>= 1) {
        if (threadIdx.x < s) { s1[threadIdx.x] += s1[threadIdx.x + s]; s2[threadIdx.x] += s2[threadIdx.x + s]; }
        __syncthreads();
    }
    float mean = s1[0] / (float)HW;
    float var  = s2[0] / (float)HW - mean * mean;
    float inv  = rsqrtf(var + 1e-5f);
    for (int i = threadIdx.x; i < HW; i += blockDim.x) {
        float v = (p[i] - mean) * inv;
        if (relu) v = fmaxf(v, 0.f);
        p[i] = v;
    }
}

// ------------------------- elementwise helpers -----------------------
__global__ void add_relu_kernel(const float* __restrict__ a, const float* __restrict__ b,
                                float* __restrict__ o, long n)
{
    long i = (long)blockIdx.x * blockDim.x + threadIdx.x;
    if (i < n) o[i] = fmaxf(a[i] + b[i], 0.f);
}
__global__ void mul_kernel(const float* __restrict__ a, const float* __restrict__ b,
                           float* __restrict__ o, long n)
{
    long i = (long)blockIdx.x * blockDim.x + threadIdx.x;
    if (i < n) o[i] = a[i] * b[i];
}
__global__ void gru_combine_kernel(float* __restrict__ h, const float* __restrict__ z,
                                   const float* __restrict__ q, long n)
{
    long i = (long)blockIdx.x * blockDim.x + threadIdx.x;
    if (i < n) { float zz = z[i]; h[i] = (1.f - zz) * h[i] + zz * q[i]; }
}
__global__ void add_inplace_kernel(float* __restrict__ a, const float* __restrict__ b, long n)
{
    long i = (long)blockIdx.x * blockDim.x + threadIdx.x;
    if (i < n) a[i] += b[i];
}
__global__ void copy_ch_kernel(const float* __restrict__ src, float* __restrict__ dst,
                               int B, int C, int HW, int dstC, int dstOff)
{
    long i = (long)blockIdx.x * blockDim.x + threadIdx.x;
    long total = (long)B * C * HW;
    if (i >= total) return;
    int hw = (int)(i % HW);
    long t = i / HW;
    int c = (int)(t % C);
    int b = (int)(t / C);
    dst[((size_t)b * dstC + dstOff + c) * HW + hw] = src[i];
}
__global__ void split_ctx_kernel(const float* __restrict__ ctx, float* __restrict__ net,
                                 float* __restrict__ inp, int B, int HW)
{
    long i = (long)blockIdx.x * blockDim.x + threadIdx.x;
    long total = (long)B * 128 * HW;
    if (i >= total) return;
    int hw = (int)(i % HW);
    long t = i / HW;
    int c = (int)(t % 128);
    int b = (int)(t / 128);
    net[i] = tanhf(ctx[((size_t)b * 256 + c) * HW + hw]);
    inp[i] = fmaxf(ctx[((size_t)b * 256 + 128 + c) * HW + hw], 0.f);
}
__global__ void avgpool2_kernel(const float* __restrict__ in, float* __restrict__ out,
                                long N, int H, int W)
{
    int Ho = H >> 1, Wo = W >> 1;
    long i = (long)blockIdx.x * blockDim.x + threadIdx.x;
    long total = N * Ho * Wo;
    if (i >= total) return;
    int xo = (int)(i % Wo);
    long t = i / Wo;
    int yo = (int)(t % Ho);
    long n = t / Ho;
    const float* p = in + (size_t)n * H * W;
    float v = p[(2 * yo) * W + 2 * xo] + p[(2 * yo) * W + 2 * xo + 1]
            + p[(2 * yo + 1) * W + 2 * xo] + p[(2 * yo + 1) * W + 2 * xo + 1];
    out[i] = 0.25f * v;
}
__global__ void coords_init_kernel(float* __restrict__ c, int B, int h, int w)
{
    long i = (long)blockIdx.x * blockDim.x + threadIdx.x;
    long total = (long)B * 2 * h * w;
    if (i >= total) return;
    int x = (int)(i % w);
    long t = i / w;
    int y = (int)(t % h);
    int ch = (int)((t / h) % 2);
    c[i] = (ch == 0) ? (float)x : (float)y;
}
__global__ void flow_from_coords_kernel(const float* __restrict__ c, float* __restrict__ f,
                                        int B, int h, int w)
{
    long i = (long)blockIdx.x * blockDim.x + threadIdx.x;
    long total = (long)B * 2 * h * w;
    if (i >= total) return;
    int x = (int)(i % w);
    long t = i / w;
    int y = (int)(t % h);
    int ch = (int)((t / h) % 2);
    f[i] = c[i] - ((ch == 0) ? (float)x : (float)y);
}

// ---------------------------------------------------------------------
// Correlation pyramid lookup: bilinear samples with zeros padding.
// ---------------------------------------------------------------------
__global__ void corr_lookup_kernel(const float* __restrict__ p0, const float* __restrict__ p1,
                                   const float* __restrict__ p2, const float* __restrict__ p3,
                                   const float* __restrict__ coords, float* __restrict__ out,
                                   int B, int h, int w)
{
    long idx = (long)blockIdx.x * blockDim.x + threadIdx.x;
    long total = (long)B * h * w * 4 * 81;
    if (idx >= total) return;
    int j = (int)(idx % 9); long t = idx / 9;
    int i = (int)(t % 9);  t /= 9;
    int l = (int)(t % 4);  t /= 4;
    int x = (int)(t % w);  t /= w;
    int y = (int)(t % h);
    int b = (int)(t / h);
    long n = ((long)b * h + y) * w + x;
    float cx = coords[(((size_t)b * 2 + 0) * h + y) * w + x];
    float cy = coords[(((size_t)b * 2 + 1) * h + y) * w + x];
    float sc = 1.f / (float)(1 << l);
    float sx = cx * sc + (float)(i - 4);
    float sy = cy * sc + (float)(j - 4);
    int Hl = h >> l, Wl = w >> l;
    const float* fm = (l == 0 ? p0 : l == 1 ? p1 : l == 2 ? p2 : p3) + (size_t)n * Hl * Wl;
    float x0 = floorf(sx), y0 = floorf(sy);
    float wx = sx - x0, wy = sy - y0;
    float v = 0.f;
#pragma unroll
    for (int dy = 0; dy < 2; ++dy)
#pragma unroll
        for (int dx = 0; dx < 2; ++dx) {
            float xi = x0 + dx, yi = y0 + dy;
            bool ok = (xi >= 0.f && xi <= (float)(Wl - 1) && yi >= 0.f && yi <= (float)(Hl - 1));
            int xq = iclamp((int)xi, 0, Wl - 1);
            int yq = iclamp((int)yi, 0, Hl - 1);
            float wt = (dx ? wx : 1.f - wx) * (dy ? wy : 1.f - wy);
            float s = fm[(size_t)yq * Wl + xq];
            v += ok ? s * wt : 0.f;
        }
    int ch = l * 81 + i * 9 + j;
    out[((size_t)(b * 324 + ch) * h + y) * w + x] = v;
}

// ---------------------------------------------------------------------
// Convex-combination 8x upsampling; softmax over 9 of 0.25*mask.
// ---------------------------------------------------------------------
__global__ void upsample_kernel(const float* __restrict__ flow, const float* __restrict__ mask,
                                float* __restrict__ out, int B, int h, int w)
{
    long idx = (long)blockIdx.x * blockDim.x + threadIdx.x;
    long total = (long)B * h * w * 64;
    if (idx >= total) return;
    int j = (int)(idx % 8); long t = idx / 8;
    int i = (int)(t % 8);  t /= 8;
    int x = (int)(t % w);  t /= w;
    int y = (int)(t % h);
    int b = (int)(t / h);

    float lg[9]; float mx = -1e30f;
#pragma unroll
    for (int k = 0; k < 9; ++k) {
        lg[k] = 0.25f * mask[((size_t)(b * 576 + (k * 8 + i) * 8 + j) * h + y) * w + x];
        mx = fmaxf(mx, lg[k]);
    }
    float den = 0.f;
#pragma unroll
    for (int k = 0; k < 9; ++k) { lg[k] = __expf(lg[k] - mx); den += lg[k]; }
#pragma unroll
    for (int c = 0; c < 2; ++c) {
        float acc = 0.f;
#pragma unroll
        for (int k = 0; k < 9; ++k) {
            int ky = k / 3, kx = k - 3 * (k / 3);
            int yy = y + ky - 1, xx = x + kx - 1;
            float f = (yy >= 0 && yy < h && xx >= 0 && xx < w)
                        ? 8.f * flow[((size_t)(b * 2 + c) * h + yy) * w + xx] : 0.f;
            acc += lg[k] * f;
        }
        out[((size_t)(b * 2 + c) * (8 * h) + (8 * y + i)) * (8 * w) + (8 * x + j)] = acc / den;
    }
}

// =====================================================================
// Host-side orchestration
// =====================================================================
struct RBP  { const float *c1w, *c1b, *c2w, *c2b, *dw, *db; };
struct EncP { const float *c1w, *c1b, *c2w, *c2b; RBP l1a, l1b, l2a, l2b, l3a, l3b; };

struct PW  { const _Float16* w; const float* b; int Cinp; };
struct RBW { PW c1, c2, dw; bool has_dw; };
struct EncW { PW c1, c2; RBW l1a, l1b, l2a, l2b, l3a, l3b; };

static inline int ewg(long n) { return (int)((n + EW_BLOCK - 1) / EW_BLOCK); }

extern "C" void kernel_launch(void* const* d_in, const int* in_sizes, int n_in,
                              void* d_out, int out_size, void* d_ws, size_t ws_size,
                              hipStream_t stream)
{
    (void)in_sizes; (void)n_in; (void)out_size; (void)ws_size;
    const int Bc = 2, H = 384, W = 512;
    const int h = H / 8, w = W / 8;     // 48 x 64
    const int Np = h * w;               // 3072
    const int ITERS = 6;                // setup_inputs iters=6 (device scalar unreadable in capture)

    const float* img1 = (const float*)d_in[0];
    const float* img2 = (const float*)d_in[1];

    // ---- parse params (jax tree_flatten: alphabetical keys per dict level)
    int cur = 2;
    auto nxt = [&]() { return (const float*)d_in[cur++]; };
    auto parse_rb = [&](bool ds) {
        RBP r{};
        r.c1b = nxt(); r.c1w = nxt(); r.c2b = nxt(); r.c2w = nxt();
        if (ds) { r.db = nxt(); r.dw = nxt(); } else { r.db = nullptr; r.dw = nullptr; }
        return r;
    };
    auto parse_enc = [&]() {
        EncP e{};
        e.c1b = nxt(); e.c1w = nxt(); e.c2b = nxt(); e.c2w = nxt();
        e.l1a = parse_rb(false); e.l1b = parse_rb(false);
        e.l2a = parse_rb(true);  e.l2b = parse_rb(false);
        e.l3a = parse_rb(true);  e.l3b = parse_rb(false);
        return e;
    };
    EncP cnetP = parse_enc();
    EncP fnetP = parse_enc();
    const float *u_cb  = nxt(), *u_cc1b = nxt(), *u_cc1w = nxt(), *u_cc2b = nxt(), *u_cc2w = nxt();
    const float *u_cf1b = nxt(), *u_cf1w = nxt(), *u_cf2b = nxt(), *u_cf2w = nxt(), *u_cw = nxt();
    const float *u_f1b = nxt(), *u_f1w = nxt(), *u_f2b = nxt(), *u_f2w = nxt();
    const float *u_m1b = nxt(), *u_m1w = nxt(), *u_m2b = nxt(), *u_m2w = nxt();
    const float *u_q1b = nxt(), *u_q1w = nxt(), *u_q2b = nxt(), *u_q2w = nxt();
    const float *u_r1b = nxt(), *u_r1w = nxt(), *u_r2b = nxt(), *u_r2w = nxt();
    const float *u_z1b = nxt(), *u_z1w = nxt(), *u_z2b = nxt(), *u_z2w = nxt();

    // ---- workspace carve (floats, 32B aligned blocks)
    float* wsf = (float*)d_ws;
    size_t off = 0;
    auto alloc = [&](size_t n) { float* p = wsf + off; off += (n + 7) & ~(size_t)7; return p; };
    const size_t encN = (size_t)Bc * 64 * (H / 2) * (W / 2);
    float* EA = alloc(encN); float* EB = alloc(encN);
    float* EC = alloc(encN); float* ED = alloc(encN);
    float* F1  = alloc((size_t)Bc * 256 * Np);
    float* F2  = alloc((size_t)Bc * 256 * Np);
    float* CTX = alloc((size_t)Bc * 256 * Np);
    float* NET = alloc((size_t)Bc * 128 * Np);
    float* INP = alloc((size_t)Bc * 128 * Np);
    float* P0 = alloc((size_t)Bc * Np * Np);
    float* P1 = alloc((size_t)Bc * Np * (Np / 4));
    float* P2 = alloc((size_t)Bc * Np * (Np / 16));
    float* P3 = alloc((size_t)Bc * Np * (Np / 64));
    float* COORDS = alloc((size_t)Bc * 2 * Np);
    float* FLOW   = alloc((size_t)Bc * 2 * Np);
    float* CORRF  = alloc((size_t)Bc * 324 * Np);
    float* COR1 = alloc((size_t)Bc * 256 * Np);
    float* COR2 = alloc((size_t)Bc * 128 * Np);
    float* FLO1 = alloc((size_t)Bc * 128 * Np);
    float* FLO2 = alloc((size_t)Bc * 64 * Np);
    float* CAT  = alloc((size_t)Bc * 192 * Np);
    float* MOT  = alloc((size_t)Bc * 128 * Np);
    float* Xb   = alloc((size_t)Bc * 258 * Np);
    float* HXb  = alloc((size_t)Bc * 386 * Np);
    float* Zb   = alloc((size_t)Bc * 128 * Np);
    float* Rb   = alloc((size_t)Bc * 128 * Np);
    float* Qb   = alloc((size_t)Bc * 128 * Np);
    float* RHb  = alloc((size_t)Bc * 128 * Np);
    float* QIN  = alloc((size_t)Bc * 386 * Np);
    float* T256 = alloc((size_t)Bc * 256 * Np);
    float* DLT  = alloc((size_t)Bc * 2 * Np);
    float* MSK  = alloc((size_t)Bc * 576 * Np);
    // f16 scratch: activation shadow (max: conv1 input 2*390*518*32 halves)
    _Float16* X16 = (_Float16*)alloc(7 * 1024 * 1024);
    _Float16* F1T = (_Float16*)alloc((size_t)Bc * Np * 256 / 2 + 8);
    _Float16* F2T = (_Float16*)alloc((size_t)Bc * Np * 256 / 2 + 8);

    // ---- pack all conv weights to f16 fragment layout (once per call)
    auto packW = [&](const float* wsrc, const float* bsrc, int Cout, int Cin, int kh, int kw) -> PW {
        int Cinp = (Cin + 31) & ~31, Cout16 = (Cout + 15) & ~15, khkw = kh * kw;
        size_t n = (size_t)Cout16 * khkw * Cinp;
        _Float16* dst = (_Float16*)alloc(n / 2 + 8);
        pack_weight_kernel<<<ewg((long)n), EW_BLOCK, 0, stream>>>(wsrc, dst, Cout, Cin, khkw, Cinp, Cout16);
        return PW{dst, bsrc, Cinp};
    };
    auto packRB = [&](const RBP& p, int Ci, int Co) -> RBW {
        RBW r{};
        r.c1 = packW(p.c1w, p.c1b, Co, Ci, 3, 3);
        r.c2 = packW(p.c2w, p.c2b, Co, Co, 3, 3);
        r.has_dw = (p.dw != nullptr);
        if (r.has_dw) r.dw = packW(p.dw, p.db, Co, Ci, 1, 1);
        return r;
    };
    auto packEnc = [&](const EncP& p) -> EncW {
        EncW e{};
        e.c1 = packW(p.c1w, p.c1b, 64, 3, 7, 7);
        e.l1a = packRB(p.l1a, 64, 64);   e.l1b = packRB(p.l1b, 64, 64);
        e.l2a = packRB(p.l2a, 64, 128);  e.l2b = packRB(p.l2b, 128, 128);
        e.l3a = packRB(p.l3a, 128, 192); e.l3b = packRB(p.l3b, 192, 192);
        e.c2 = packW(p.c2w, p.c2b, 256, 192, 3, 3);
        return e;
    };
    EncW cnet = packEnc(cnetP);
    EncW fnet = packEnc(fnetP);
    PW cc1 = packW(u_cc1w, u_cc1b, 256, 324, 1, 1);
    PW cc2 = packW(u_cc2w, u_cc2b, 128, 256, 3, 3);
    PW cf1 = packW(u_cf1w, u_cf1b, 128, 2, 7, 7);
    PW cf2 = packW(u_cf2w, u_cf2b, 64, 128, 3, 3);
    PW cwp = packW(u_cw,  u_cb,  128, 192, 3, 3);
    PW z1 = packW(u_z1w, u_z1b, 128, 386, 1, 5);
    PW r1 = packW(u_r1w, u_r1b, 128, 386, 1, 5);
    PW q1 = packW(u_q1w, u_q1b, 128, 386, 1, 5);
    PW z2 = packW(u_z2w, u_z2b, 128, 386, 5, 1);
    PW r2 = packW(u_r2w, u_r2b, 128, 386, 5, 1);
    PW q2 = packW(u_q2w, u_q2b, 128, 386, 5, 1);
    PW f1p = packW(u_f1w, u_f1b, 256, 128, 3, 3);
    PW f2p = packW(u_f2w, u_f2b, 2, 256, 1, 1);
    PW m1p = packW(u_m1w, u_m1b, 256, 128, 3, 3);
    PW m2p = packW(u_m2w, u_m2b, 576, 256, 1, 1);

    // ---- conv launcher: shadow conversion + WMMA conv
    auto conv = [&](const float* x, const PW& pw, float* y,
                    int Cin, int Hin, int Win, int Cout,
                    int kh, int kw, int stride, int act,
                    int* HoutOut = (int*)nullptr, int* WoutOut = (int*)nullptr) {
        int Hout = (Hin + 2 * (kh / 2) - kh) / stride + 1;
        int Wout = (Win + 2 * (kw / 2) - kw) / stride + 1;
        if (HoutOut) *HoutOut = Hout;
        if (WoutOut) *WoutOut = Wout;
        long ns = (long)Bc * (Hin + 2 * HALO) * (Win + 2 * HALO) * pw.Cinp;
        shadow_kernel<<<ewg(ns), EW_BLOCK, 0, stream>>>(x, X16, Bc, Cin, pw.Cinp, Hin, Win);
        int Npix = Hout * Wout;
        int tiles = Bc * ((Cout + 15) / 16) * ((Npix + 31) / 32);
        int wpb = 8;
        int blocks = (tiles + wpb - 1) / wpb;
        dim3 bs(32 * wpb);
#define CONV_CASE(KH, KW)                                                               \
        conv2d_wmma_kernel<KH, KW><<<blocks, bs, 0, stream>>>(X16, pw.w, pw.b, y, Bc,   \
            pw.Cinp, Hin, Win, Cout, Hout, Wout, stride, act)
        if      (kh == 1 && kw == 1) CONV_CASE(1, 1);
        else if (kh == 3 && kw == 3) CONV_CASE(3, 3);
        else if (kh == 7 && kw == 7) CONV_CASE(7, 7);
        else if (kh == 1 && kw == 5) CONV_CASE(1, 5);
        else                         CONV_CASE(5, 1);
#undef CONV_CASE
    };
    auto inorm = [&](float* d, int planes, int HW, int relu) {
        instnorm_kernel<<<planes, EW_BLOCK, 0, stream>>>(d, HW, relu);
    };
    auto res_block = [&](const float* x, float* out, float* t1, float* t2,
                         const RBW& p, int Ci, int Co, int Hi, int Wi, int stride) {
        int Ho, Wo;
        conv(x, p.c1, t1, Ci, Hi, Wi, Co, 3, 3, stride, 0, &Ho, &Wo);
        inorm(t1, Bc * Co, Ho * Wo, 1);
        conv(t1, p.c2, t2, Co, Ho, Wo, Co, 3, 3, 1, 0);
        inorm(t2, Bc * Co, Ho * Wo, 1);
        long n = (long)Bc * Co * Ho * Wo;
        if (p.has_dw) {
            conv(x, p.dw, t1, Ci, Hi, Wi, Co, 1, 1, stride, 0);
            inorm(t1, Bc * Co, Ho * Wo, 0);
            add_relu_kernel<<<ewg(n), EW_BLOCK, 0, stream>>>(t1, t2, out, n);
        } else {
            add_relu_kernel<<<ewg(n), EW_BLOCK, 0, stream>>>(x, t2, out, n);
        }
    };
    auto encoder = [&](const float* img, const EncW& p, float* out) {
        int H2 = H / 2, W2 = W / 2;
        conv(img, p.c1, EA, 3, H, W, 64, 7, 7, 2, 0);
        inorm(EA, Bc * 64, H2 * W2, 1);
        res_block(EA, EB, EC, ED, p.l1a, 64, 64, H2, W2, 1);
        res_block(EB, EA, EC, ED, p.l1b, 64, 64, H2, W2, 1);
        res_block(EA, EB, EC, ED, p.l2a, 64, 128, H2, W2, 2);
        res_block(EB, EA, EC, ED, p.l2b, 128, 128, H2 / 2, W2 / 2, 1);
        res_block(EA, EB, EC, ED, p.l3a, 128, 192, H2 / 2, W2 / 2, 2);
        res_block(EB, EA, EC, ED, p.l3b, 192, 192, H2 / 4, W2 / 4, 1);
        conv(EA, p.c2, out, 192, H2 / 4, W2 / 4, 256, 3, 3, 1, 0);
    };

    // ---- encoders
    encoder(img1, fnet, F1);
    encoder(img2, fnet, F2);
    encoder(img1, cnet, CTX);
    split_ctx_kernel<<<ewg((long)Bc * 128 * Np), EW_BLOCK, 0, stream>>>(CTX, NET, INP, Bc, Np);

    // ---- correlation pyramid (transposed f16 features)
    {
        chw_to_hwc_f16_kernel<<<ewg((long)Bc * Np * 256), EW_BLOCK, 0, stream>>>(F1, F1T, Bc, 256, Np);
        chw_to_hwc_f16_kernel<<<ewg((long)Bc * Np * 256), EW_BLOCK, 0, stream>>>(F2, F2T, Bc, 256, Np);
        int tiles = Bc * (Np / 16) * (Np / 32), wpb = 8;
        corr_wmma_kernel<<<(tiles + wpb - 1) / wpb, 32 * wpb, 0, stream>>>(
            F1T, F2T, P0, Bc, 256, Np, 1.f / 16.f);
        long N = (long)Bc * Np;
        avgpool2_kernel<<<ewg(N * (Np / 4)),  EW_BLOCK, 0, stream>>>(P0, P1, N, h,     w);
        avgpool2_kernel<<<ewg(N * (Np / 16)), EW_BLOCK, 0, stream>>>(P1, P2, N, h / 2, w / 2);
        avgpool2_kernel<<<ewg(N * (Np / 64)), EW_BLOCK, 0, stream>>>(P2, P3, N, h / 4, w / 4);
    }

    coords_init_kernel<<<ewg((long)Bc * 2 * Np), EW_BLOCK, 0, stream>>>(COORDS, Bc, h, w);
    const long n128 = (long)Bc * 128 * Np;
    const long n2   = (long)Bc * 2 * Np;

    for (int it = 0; it < ITERS; ++it) {
        corr_lookup_kernel<<<ewg((long)Bc * Np * 4 * 81), EW_BLOCK, 0, stream>>>(
            P0, P1, P2, P3, COORDS, CORRF, Bc, h, w);
        flow_from_coords_kernel<<<ewg(n2), EW_BLOCK, 0, stream>>>(COORDS, FLOW, Bc, h, w);

        // motion encoder
        conv(CORRF, cc1, COR1, 324, h, w, 256, 1, 1, 1, 1);
        conv(COR1,  cc2, COR2, 256, h, w, 128, 3, 3, 1, 1);
        conv(FLOW,  cf1, FLO1, 2,   h, w, 128, 7, 7, 1, 1);
        conv(FLO1,  cf2, FLO2, 128, h, w, 64,  3, 3, 1, 1);
        copy_ch_kernel<<<ewg(n128), EW_BLOCK, 0, stream>>>(COR2, CAT, Bc, 128, Np, 192, 0);
        copy_ch_kernel<<<ewg((long)Bc * 64 * Np), EW_BLOCK, 0, stream>>>(FLO2, CAT, Bc, 64, Np, 192, 128);
        conv(CAT, cwp, MOT, 192, h, w, 128, 3, 3, 1, 1);

        // x = concat(inp, m, flow)
        copy_ch_kernel<<<ewg(n128), EW_BLOCK, 0, stream>>>(INP, Xb, Bc, 128, Np, 258, 0);
        copy_ch_kernel<<<ewg(n128), EW_BLOCK, 0, stream>>>(MOT, Xb, Bc, 128, Np, 258, 128);
        copy_ch_kernel<<<ewg(n2),   EW_BLOCK, 0, stream>>>(FLOW, Xb, Bc, 2, Np, 258, 256);

        // two-pass separable GRU
        for (int s = 0; s < 2; ++s) {
            int kh = (s == 0) ? 1 : 5, kw = (s == 0) ? 5 : 1;
            const PW& zw = s ? z2 : z1;
            const PW& rw = s ? r2 : r1;
            const PW& qw = s ? q2 : q1;
            copy_ch_kernel<<<ewg(n128), EW_BLOCK, 0, stream>>>(NET, HXb, Bc, 128, Np, 386, 0);
            copy_ch_kernel<<<ewg((long)Bc * 258 * Np), EW_BLOCK, 0, stream>>>(Xb, HXb, Bc, 258, Np, 386, 128);
            conv(HXb, zw, Zb, 386, h, w, 128, kh, kw, 1, 2);  // sigmoid
            conv(HXb, rw, Rb, 386, h, w, 128, kh, kw, 1, 2);  // sigmoid
            mul_kernel<<<ewg(n128), EW_BLOCK, 0, stream>>>(Rb, NET, RHb, n128);
            copy_ch_kernel<<<ewg(n128), EW_BLOCK, 0, stream>>>(RHb, QIN, Bc, 128, Np, 386, 0);
            copy_ch_kernel<<<ewg((long)Bc * 258 * Np), EW_BLOCK, 0, stream>>>(Xb, QIN, Bc, 258, Np, 386, 128);
            conv(QIN, qw, Qb, 386, h, w, 128, kh, kw, 1, 3);  // tanh
            gru_combine_kernel<<<ewg(n128), EW_BLOCK, 0, stream>>>(NET, Zb, Qb, n128);
        }

        // flow head + mask head
        conv(NET,  f1p, T256, 128, h, w, 256, 3, 3, 1, 1);
        conv(T256, f2p, DLT,  256, h, w, 2,   1, 1, 1, 0);
        conv(NET,  m1p, T256, 128, h, w, 256, 3, 3, 1, 1);
        conv(T256, m2p, MSK,  256, h, w, 576, 1, 1, 1, 0);

        add_inplace_kernel<<<ewg(n2), EW_BLOCK, 0, stream>>>(COORDS, DLT, n2);

        if (it == ITERS - 1) {
            flow_from_coords_kernel<<<ewg(n2), EW_BLOCK, 0, stream>>>(COORDS, FLOW, Bc, h, w);
            upsample_kernel<<<ewg((long)Bc * Np * 64), EW_BLOCK, 0, stream>>>(
                FLOW, MSK, (float*)d_out, Bc, h, w);
        }
    }
}
